// EdgeConvfeat_36490042147211
// MI455X (gfx1250) — matrix operand order repaired and checked by reading, not compiled, not run on hardware
//
#include <hip/hip_runtime.h>
#include <cstdint>
#include <cstddef>

// ---------------------------------------------------------------------------
// CDNA5 (gfx1250) implementation of the RigNet-style GNN forward pass.
// All GEMMs run on v_wmma_f32_16x16x32_f16 (f16 in, f32 accumulate).
// Max-aggregation uses uint atomicMax on float bits (values are post-ReLU >=0).
// ---------------------------------------------------------------------------

typedef _Float16 h16 __attribute__((ext_vector_type(16)));
typedef float    f32x8 __attribute__((ext_vector_type(8)));
typedef unsigned int u32x4 __attribute__((ext_vector_type(4)));

union FragU { u32x4 u[2]; h16 h; };

// A-fragment (16x32 f16, M x K) from LDS tile stored row-major [16][ldk].
// Per ISA 7.12.2: lane m=l&15, half=l>>4; VGPRs 0-3 hold K = 8*half + {0..7},
// VGPRs 4-7 hold K = 16 + 8*half + {0..7}  -> two contiguous 16-byte chunks.
__device__ __forceinline__ h16 a_frag(const _Float16* tile, int ldk, int kbase) {
  int lane = threadIdx.x & 31;
  int m = lane & 15, hf = lane >> 4;
  const _Float16* p = tile + m * ldk + kbase + 8 * hf;
  FragU f;
  f.u[0] = *(const u32x4*)p;
  f.u[1] = *(const u32x4*)(p + 16);
  return f.h;
}

// B-fragment (32x16 f16, K x N): 16 contiguous halves per lane from the
// pre-packed weight buffer; p already includes the lane offset.
__device__ __forceinline__ h16 b_frag_at(const _Float16* p) {
  FragU f;
  f.u[0] = *(const u32x4*)p;
  f.u[1] = *(const u32x4*)(p + 8);
  return f.h;
}

__device__ __forceinline__ f32x8 wmma_f16(h16 a, h16 b, f32x8 c) {
  return __builtin_amdgcn_wmma_f32_16x16x32_f16(false, a, false, b, (short)0, c,
                                                false, false);
}

// ---------------------------------------------------------------------------
// Weight packer: f32 [Kact][Oact] row-major -> f16 WMMA B-fragment order,
// zero-padded to [Kpad][Opad]. For tile (kt,nt), lane l's 16 halves live at
// ((kt*NT+nt)*32 + l)*16.  Within a lane: half i holds K = 8*(l>>4) + (i&7)
// + 16*(i>>3), N = lane&15.
// ---------------------------------------------------------------------------
__global__ void pack_w_kernel(const float* __restrict__ src, _Float16* __restrict__ dst,
                              int Kact, int Oact, int Kpad, int Opad) {
  int idx = blockIdx.x * blockDim.x + threadIdx.x;
  int total = Kpad * Opad;
  if (idx >= total) return;
  int blk = idx >> 9, w = idx & 511;
  int lane = w >> 4, i = w & 15;
  int NT = Opad >> 4;
  int kt = blk / NT, nt = blk - kt * NT;
  int hf = lane >> 4, ncl = lane & 15;
  int v = i >> 1;
  int kl = 8 * hf + (i & 7) + 16 * (i >> 3);
  int gk = kt * 32 + kl, gn = nt * 16 + ncl;
  float val = (gk < Kact && gn < Oact) ? src[(size_t)gk * Oact + gn] : 0.f;
  dst[idx] = (_Float16)val;
}

// x0 = pad(concat(pos, vg)) to 32 channels, f16.
__global__ void prep_x0_kernel(const float* __restrict__ pos, const float* __restrict__ vg,
                               _Float16* __restrict__ x0, int n) {
  int idx = blockIdx.x * blockDim.x + threadIdx.x;
  if (idx >= n * 32) return;
  int row = idx >> 5, c = idx & 31;
  float v = 0.f;
  if (c < 3) v = pos[row * 3 + c];
  else if (c < 29) v = vg[row * 26 + (c - 3)];
  x0[idx] = (_Float16)v;
}

__global__ void zero_kernel(float* __restrict__ p, long long n) {
  long long i = (long long)blockIdx.x * blockDim.x + threadIdx.x;
  if (i < n) p[i] = 0.f;
}

// Per-batch segment max over x4 (f16), writes xg (f32) and the output tail.
__global__ void segmax_kernel(const _Float16* __restrict__ x4, float* __restrict__ xg,
                              float* __restrict__ outtail, int npg, int C, int total) {
  int idx = blockIdx.x * blockDim.x + threadIdx.x;
  if (idx >= total) return;
  int b = idx / C, c = idx - b * C;
  const _Float16* p = x4 + (size_t)b * npg * C + c;
  float m = 0.f;  // x4 is post-ReLU >= 0
  for (int r = 0; r < npg; ++r) m = fmaxf(m, (float)p[(size_t)r * C]);
  xg[idx] = m;
  outtail[idx] = m;
}

// ---------------------------------------------------------------------------
// EdgeConv: per wave, 16 edges. Stage A=[x_dst, x_src-x_dst] (f16) in LDS,
// GEMM1 (Kpad -> H) + bias + ReLU into LDS, GEMM2 (H -> H) + bias + ReLU,
// scatter-max into agg[dst][coloff + n] via uint atomicMax.
// All B-fragment addresses are constant offsets from one per-lane base.
// ---------------------------------------------------------------------------
template <int KPAD, int H, int WAVES>
__global__ __launch_bounds__(WAVES * 32) void edgeconv_kernel(
    const _Float16* __restrict__ xh, int ldx, int cin,
    const int* __restrict__ ei, int E,
    const _Float16* __restrict__ w1, const float* __restrict__ b1,
    const _Float16* __restrict__ w2, const float* __restrict__ b2,
    float* __restrict__ agg, int ldagg, int coloff) {
  __shared__ alignas(16) _Float16 At[WAVES][16 * KPAD];
  __shared__ alignas(16) _Float16 Mt[WAVES][16 * H];
  const int wid = threadIdx.x >> 5, lane = threadIdx.x & 31;
  const int ebase = (blockIdx.x * WAVES + wid) * 16;
  _Float16* A = At[wid];
  _Float16* M = Mt[wid];
  const int ncl = lane & 15, hf = lane >> 4;
  constexpr int NT1 = H / 16;

  // ---- stage A rows into LDS (row-outer: one ei load pair per row) ----
#pragma unroll 4
  for (int e = 0; e < 16; ++e) {
    int src = ei[ebase + e], dst = ei[E + ebase + e];
    const _Float16* xs = xh + (size_t)src * ldx;
    const _Float16* xd = xh + (size_t)dst * ldx;
#pragma unroll
    for (int k = lane; k < KPAD; k += 32) {
      float v = 0.f;
      if (k < cin) {
        v = (float)xd[k];
      } else if (k < 2 * cin) {
        int c = k - cin;
        v = (float)xs[c] - (float)xd[c];
      }
      A[e * KPAD + k] = (_Float16)v;
    }
  }
  __syncthreads();

  // ---- GEMM1: A[16xKPAD] x W1[KPADxH] -> M (ReLU, f16, in LDS) ----
  for (int nt = 0; nt < NT1; ++nt) {
    f32x8 c = {};
    const _Float16* bp = w1 + ((size_t)nt * 32 + lane) * 16;
#pragma unroll
    for (int kt = 0; kt < KPAD / 32; ++kt)
      c = wmma_f16(a_frag(A, KPAD, kt * 32), b_frag_at(bp + (size_t)kt * (NT1 * 512)), c);
    float bias = b1[nt * 16 + ncl];
#pragma unroll
    for (int r = 0; r < 8; ++r) {
      float v = c[r] + bias;
      M[(r + 8 * hf) * H + nt * 16 + ncl] = (_Float16)(v > 0.f ? v : 0.f);
    }
  }
  __syncthreads();

  // ---- GEMM2: M[16xH] x W2[HxH] -> scatter-max into agg ----
  int dstv[8];
#pragma unroll
  for (int r = 0; r < 8; ++r) dstv[r] = ei[E + ebase + r + 8 * hf];
  float* aggrow[8];
#pragma unroll
  for (int r = 0; r < 8; ++r) aggrow[r] = agg + (size_t)dstv[r] * ldagg + coloff + ncl;

  for (int nt = 0; nt < NT1; ++nt) {
    f32x8 c = {};
    const _Float16* bp = w2 + ((size_t)nt * 32 + lane) * 16;
#pragma unroll
    for (int kt = 0; kt < H / 32; ++kt)
      c = wmma_f16(a_frag(M, H, kt * 32), b_frag_at(bp + (size_t)kt * (NT1 * 512)), c);
    float bias = b2[nt * 16 + ncl];
#pragma unroll
    for (int r = 0; r < 8; ++r) {
      float v = c[r] + bias;
      if (v > 0.f)  // max against 0-initialized agg; <=0 contributes nothing
        atomicMax((unsigned int*)(aggrow[r] + nt * 16), __float_as_uint(v));
    }
  }
}

// ---------------------------------------------------------------------------
// Dense layer with gather/concat of up to 5 channel segments.
// Block = 256 threads (8 waves) x 16 rows; wave `wid` owns tiles nt = wid+8t.
// K streamed via 16x64 f16 LDS chunk. wid is forced into an SGPR so tile
// guards are scalar branches (EXEC stays full for WMMA); the EXACT variant
// (NT == 8*MAXT) has no guards at all.
// ---------------------------------------------------------------------------
struct Segs5 {
  const void* p[5];
  int stride[5];
  int nch[5];
  int rowdiv[5];  // 0 = identity, else source row = row / rowdiv (xg gather)
  int isf16[5];   // 1 = f16 source, 0 = f32 source
  int nseg;
};

template <int MAXT, bool EXACT>
__global__ __launch_bounds__(256) void dense_kernel(
    Segs5 segs, int K, const _Float16* __restrict__ wp, const float* __restrict__ bias,
    int O, int Oact, int relu, void* __restrict__ out, int ldout, int out_f32,
    int nrows) {
  __shared__ alignas(16) _Float16 Ac[16 * 64];
  const int lane = threadIdx.x & 31;
  const int wid = __builtin_amdgcn_readfirstlane(threadIdx.x >> 5);
  const int rbase = blockIdx.x * 16;
  const int NT = O >> 4;
  const int myNT = EXACT ? MAXT : ((NT - wid + 7) >> 3);
  f32x8 acc[MAXT] = {};

  const int nchunk = K >> 6;
  for (int kc = 0; kc < nchunk; ++kc) {
    // ---- stage 16x64 A chunk (f16) with segment gather/concat ----
    for (int t = threadIdx.x; t < 16 * 64; t += 256) {
      int r = t >> 6, k = t & 63;
      int gk = kc * 64 + k;
      int row = rbase + r;
      float v = 0.f;
      if (row < nrows) {
        int off = gk;
        for (int s = 0; s < segs.nseg; ++s) {
          if (off < segs.nch[s]) {
            int rr = segs.rowdiv[s] ? row / segs.rowdiv[s] : row;
            size_t ix = (size_t)rr * segs.stride[s] + off;
            v = segs.isf16[s] ? (float)((const _Float16*)segs.p[s])[ix]
                              : ((const float*)segs.p[s])[ix];
            break;
          }
          off -= segs.nch[s];
        }
      }
      Ac[t] = (_Float16)v;
    }
    __syncthreads();
    // One per-lane base pointer; every fragment is a constant offset from it.
    const _Float16* bp = wp + ((size_t)(kc * 2) * NT + wid) * 512 + lane * 16;
#pragma unroll
    for (int kk = 0; kk < 2; ++kk) {
      h16 a = a_frag(Ac, 64, kk * 32);
#pragma unroll
      for (int t = 0; t < MAXT; ++t) {
        if (!EXACT && t >= myNT) break;  // scalar branch (wid, NT uniform)
        h16 b = b_frag_at(bp + (size_t)kk * NT * 512 + t * 4096);
        acc[t] = wmma_f16(a, b, acc[t]);
      }
    }
    __syncthreads();
  }

  const int ncl = lane & 15, hf = lane >> 4;
#pragma unroll
  for (int t = 0; t < MAXT; ++t) {
    if (!EXACT && t >= myNT) break;
    int nt = wid + t * 8;
    int n = nt * 16 + ncl;
    float bv = (n < Oact) ? bias[n] : 0.f;
#pragma unroll
    for (int r = 0; r < 8; ++r) {
      int row = rbase + r + 8 * hf;
      if (row < nrows && n < Oact) {
        float v = acc[t][r] + bv;
        if (relu && v < 0.f) v = 0.f;
        if (out_f32) ((float*)out)[(size_t)row * ldout + n] = v;
        else ((_Float16*)out)[(size_t)row * ldout + n] = (_Float16)v;
      }
    }
  }
}

static void launch_dense(const Segs5& s, int K, const _Float16* wp, const float* bias,
                         int O, int Oact, int relu, void* out, int ldout, int out_f32,
                         int nrows, hipStream_t stream) {
  dim3 g((nrows + 15) / 16), b(256);
  int NT = O / 16, maxt = (NT + 7) / 8;
  bool exact = (NT == maxt * 8);
  if (maxt <= 1) {
    if (exact) dense_kernel<1, true ><<<g, b, 0, stream>>>(s, K, wp, bias, O, Oact, relu, out, ldout, out_f32, nrows);
    else       dense_kernel<1, false><<<g, b, 0, stream>>>(s, K, wp, bias, O, Oact, relu, out, ldout, out_f32, nrows);
  } else if (maxt <= 2) {
    if (exact) dense_kernel<2, true ><<<g, b, 0, stream>>>(s, K, wp, bias, O, Oact, relu, out, ldout, out_f32, nrows);
    else       dense_kernel<2, false><<<g, b, 0, stream>>>(s, K, wp, bias, O, Oact, relu, out, ldout, out_f32, nrows);
  } else if (maxt <= 4) {
    if (exact) dense_kernel<4, true ><<<g, b, 0, stream>>>(s, K, wp, bias, O, Oact, relu, out, ldout, out_f32, nrows);
    else       dense_kernel<4, false><<<g, b, 0, stream>>>(s, K, wp, bias, O, Oact, relu, out, ldout, out_f32, nrows);
  } else {
    if (exact) dense_kernel<8, true ><<<g, b, 0, stream>>>(s, K, wp, bias, O, Oact, relu, out, ldout, out_f32, nrows);
    else       dense_kernel<8, false><<<g, b, 0, stream>>>(s, K, wp, bias, O, Oact, relu, out, ldout, out_f32, nrows);
  }
}

// ---------------------------------------------------------------------------
// Orchestration. Params assumed flattened in setup_inputs() insertion order:
// pos, vg, edge_index, euc_edge_index, batch, then for each linear (g0_tpl1,
// g0_tpl2, g0_euc1, g0_euc2, g0_mlp, g1_*, g2_*, glb, t1, t2, t3): w, b.
// Workspace requirement: ~200 MB.
// ---------------------------------------------------------------------------
extern "C" void kernel_launch(void* const* d_in, const int* in_sizes, int n_in,
                              void* d_out, int out_size, void* d_ws, size_t ws_size,
                              hipStream_t stream) {
  (void)n_in; (void)ws_size;
  const float* pos = (const float*)d_in[0];
  const float* vg  = (const float*)d_in[1];
  const int* etpl  = (const int*)d_in[2];
  const int* eeuc  = (const int*)d_in[3];
  const int N   = in_sizes[0] / 3;
  const int E   = in_sizes[2] / 2;
  const int B   = (out_size - N * 72) / 1024;
  const int NPG = N / B;

  const float* W[19];
  const float* Bb[19];
  for (int i = 0; i < 19; ++i) {
    W[i]  = (const float*)d_in[5 + 2 * i];
    Bb[i] = (const float*)d_in[5 + 2 * i + 1];
  }

  static const int KA[19] = {58, 32, 58, 32, 64, 128, 128, 128, 128, 256,
                             512, 256, 512, 256, 512, 832, 1885, 1024, 256};
  static const int OA[19] = {32, 32, 32, 32, 64, 128, 128, 128, 128, 256,
                             256, 256, 256, 256, 512, 1024, 1024, 256, 72};

  char* ws = (char*)d_ws;
  size_t cur = 0;
  auto bump = [&](size_t bytes) -> void* {
    void* r = ws + cur;
    cur = (cur + bytes + 255) & ~(size_t)255;
    return r;
  };

  _Float16* wp[19];
  int KP[19], OP[19];
  for (int i = 0; i < 19; ++i) {
    KP[i] = (KA[i] + 63) & ~63;
    OP[i] = (OA[i] + 15) & ~15;
    wp[i] = (_Float16*)bump((size_t)KP[i] * OP[i] * sizeof(_Float16));
  }
  _Float16* x0h = (_Float16*)bump((size_t)N * 32 * 2);
  _Float16* x1h = (_Float16*)bump((size_t)N * 64 * 2);
  _Float16* x2h = (_Float16*)bump((size_t)N * 256 * 2);
  _Float16* x3h = (_Float16*)bump((size_t)N * 512 * 2);
  _Float16* x4h = (_Float16*)bump((size_t)N * 1024 * 2);  // glb out; reused as t1 out
  _Float16* h2h = (_Float16*)bump((size_t)N * 256 * 2);
  float* agg = (float*)bump((size_t)N * 512 * 4);
  float* xg  = (float*)bump((size_t)B * 1024 * 4);

  // Pack all weights into WMMA B-fragment order (f16).
  for (int i = 0; i < 19; ++i) {
    int total = KP[i] * OP[i];
    pack_w_kernel<<<(total + 255) / 256, 256, 0, stream>>>(W[i], wp[i], KA[i], OA[i], KP[i], OP[i]);
  }
  prep_x0_kernel<<<(N * 32 + 255) / 256, 256, 0, stream>>>(pos, vg, x0h, N);

  // ---- GCU layer 0: cin=29 (Kpad 64), h=32, co=64 ----
  zero_kernel<<<(unsigned)(((long long)N * 64 + 255) / 256), 256, 0, stream>>>(agg, (long long)N * 64);
  edgeconv_kernel<64, 32, 4><<<E / 64, 128, 0, stream>>>(x0h, 32, 29, etpl, E, wp[0], Bb[0], wp[1], Bb[1], agg, 64, 0);
  edgeconv_kernel<64, 32, 4><<<E / 64, 128, 0, stream>>>(x0h, 32, 29, eeuc, E, wp[2], Bb[2], wp[3], Bb[3], agg, 64, 32);
  {
    Segs5 s = {}; s.nseg = 1;
    s.p[0] = agg; s.stride[0] = 64; s.nch[0] = 64; s.rowdiv[0] = 0; s.isf16[0] = 0;
    launch_dense(s, KP[4], wp[4], Bb[4], OP[4], 64, 1, x1h, 64, 0, N, stream);
  }

  // ---- GCU layer 1: cin=64 (Kpad 128), h=128, co=256 ----
  zero_kernel<<<(unsigned)(((long long)N * 256 + 255) / 256), 256, 0, stream>>>(agg, (long long)N * 256);
  edgeconv_kernel<128, 128, 4><<<E / 64, 128, 0, stream>>>(x1h, 64, 64, etpl, E, wp[5], Bb[5], wp[6], Bb[6], agg, 256, 0);
  edgeconv_kernel<128, 128, 4><<<E / 64, 128, 0, stream>>>(x1h, 64, 64, eeuc, E, wp[7], Bb[7], wp[8], Bb[8], agg, 256, 128);
  {
    Segs5 s = {}; s.nseg = 1;
    s.p[0] = agg; s.stride[0] = 256; s.nch[0] = 256; s.rowdiv[0] = 0; s.isf16[0] = 0;
    launch_dense(s, KP[9], wp[9], Bb[9], OP[9], 256, 1, x2h, 256, 0, N, stream);
  }

  // ---- GCU layer 2: cin=256 (Kpad 512), h=256, co=512 ----
  zero_kernel<<<(unsigned)(((long long)N * 512 + 255) / 256), 256, 0, stream>>>(agg, (long long)N * 512);
  edgeconv_kernel<512, 256, 2><<<E / 32, 64, 0, stream>>>(x2h, 256, 256, etpl, E, wp[10], Bb[10], wp[11], Bb[11], agg, 512, 0);
  edgeconv_kernel<512, 256, 2><<<E / 32, 64, 0, stream>>>(x2h, 256, 256, eeuc, E, wp[12], Bb[12], wp[13], Bb[13], agg, 512, 256);
  {
    Segs5 s = {}; s.nseg = 1;
    s.p[0] = agg; s.stride[0] = 512; s.nch[0] = 512; s.rowdiv[0] = 0; s.isf16[0] = 0;
    launch_dense(s, KP[14], wp[14], Bb[14], OP[14], 512, 1, x3h, 512, 0, N, stream);
  }

  // ---- glb: concat(x1,x2,x3) [N,832] -> x4 [N,1024] ----
  {
    Segs5 s = {}; s.nseg = 3;
    s.p[0] = x1h; s.stride[0] = 64;  s.nch[0] = 64;  s.isf16[0] = 1;
    s.p[1] = x2h; s.stride[1] = 256; s.nch[1] = 256; s.isf16[1] = 1;
    s.p[2] = x3h; s.stride[2] = 512; s.nch[2] = 512; s.isf16[2] = 1;
    launch_dense(s, KP[15], wp[15], Bb[15], OP[15], 1024, 1, x4h, 1024, 0, N, stream);
  }

  // ---- xg = segment_max(x4, batch); also written to output tail ----
  segmax_kernel<<<(B * 1024 + 255) / 256, 256, 0, stream>>>(
      x4h, xg, (float*)d_out + (size_t)N * 72, NPG, 1024, B * 1024);

  // ---- t1: concat(xg[batch], x, x1, x2, x3) [N,1885] -> h1 [N,1024] ----
  {
    Segs5 s = {}; s.nseg = 5;
    s.p[0] = xg;  s.stride[0] = 1024; s.nch[0] = 1024; s.rowdiv[0] = NPG; s.isf16[0] = 0;
    s.p[1] = x0h; s.stride[1] = 32;   s.nch[1] = 29;   s.isf16[1] = 1;
    s.p[2] = x1h; s.stride[2] = 64;   s.nch[2] = 64;   s.isf16[2] = 1;
    s.p[3] = x2h; s.stride[3] = 256;  s.nch[3] = 256;  s.isf16[3] = 1;
    s.p[4] = x3h; s.stride[4] = 512;  s.nch[4] = 512;  s.isf16[4] = 1;
    launch_dense(s, KP[16], wp[16], Bb[16], OP[16], 1024, 1, x4h /*h1 reuse*/, 1024, 0, N, stream);
  }

  // ---- t2: h1 [N,1024] -> h2 [N,256] ----
  {
    Segs5 s = {}; s.nseg = 1;
    s.p[0] = x4h; s.stride[0] = 1024; s.nch[0] = 1024; s.isf16[0] = 1;
    launch_dense(s, KP[17], wp[17], Bb[17], OP[17], 256, 1, h2h, 256, 0, N, stream);
  }

  // ---- t3: h2 [N,256] -> out [N,72] (no relu, f32 to d_out) ----
  {
    Segs5 s = {}; s.nseg = 1;
    s.p[0] = h2h; s.stride[0] = 256; s.nch[0] = 256; s.isf16[0] = 1;
    launch_dense(s, KP[18], wp[18], Bb[18], OP[18], 72, 0, d_out, 72, 1, N, stream);
  }
}
